// CompensationAlignmentModule_12163347383101
// MI455X (gfx1250) — matrix operand order repaired and checked by reading, not compiled, not run on hardware
//
#include <hip/hip_runtime.h>
#include <hip/hip_bf16.h>
#include <math.h>

// ---------------------------------------------------------------------------
// Problem constants
// ---------------------------------------------------------------------------
#define HDIM   768
#define NROWS  8192            // B*T
#define VPOOL  1024            // token pool
#define SSAMP  256             // sampled rows
#define KTOP   64
#define TEMP_INV 10.0f         // 1/0.1
#define LIM    0.25f
#define MW     0.1f

typedef __attribute__((ext_vector_type(16))) __bf16 v16bf;
typedef __attribute__((ext_vector_type(8)))  float  v8f;
typedef int v4i __attribute__((__vector_size__(16)));

// ---------------------------------------------------------------------------
// CDNA5 async global->LDS staging (ASYNCcnt path), with safe fallback
// ---------------------------------------------------------------------------
#if __has_builtin(__builtin_amdgcn_global_load_async_to_lds_b128) && \
    __has_builtin(__builtin_amdgcn_s_wait_asynccnt)
#define USE_ASYNC_LDS 1
#else
#define USE_ASYNC_LDS 0
#endif

__device__ inline void lds_copy16(const void* g, void* l) {
#if USE_ASYNC_LDS
    __builtin_amdgcn_global_load_async_to_lds_b128(
        (__attribute__((address_space(1))) v4i*)(g),
        (__attribute__((address_space(3))) v4i*)(l), 0, 0);
#else
    *(uint4*)l = *(const uint4*)g;
#endif
}

__device__ inline void lds_copy_wait() {
#if USE_ASYNC_LDS
    __builtin_amdgcn_s_wait_asynccnt(0);
#endif
}

// ---------------------------------------------------------------------------
// Workspace layout (bytes)
// ---------------------------------------------------------------------------
static constexpr size_t NHs      = (size_t)NROWS * HDIM;
static constexpr size_t OFF_XN   = 0;                          // bf16 [N,H] normalized input
static constexpr size_t OFF_H1   = OFF_XN  + NHs * 2;          // bf16 [N,H] hidden
static constexpr size_t OFF_WB   = OFF_H1  + NHs * 2;          // bf16 [H,H] weight
static constexpr size_t OFF_HA   = OFF_WB  + (size_t)HDIM*HDIM*2; // f32 [N,H] head out A / uA
static constexpr size_t OFF_HB   = OFF_HA  + NHs * 4;          // f32 [N,H] head out B / uB
static constexpr size_t OFF_VA   = OFF_HB  + NHs * 4;          // bf16 [N,H] l2-normed view a
static constexpr size_t OFF_VB   = OFF_VA  + NHs * 2;          // bf16 [N,H] l2-normed view b
static constexpr size_t OFF_ROW  = OFF_VB  + NHs * 2;          // f32 [N] row expsum
static constexpr size_t OFF_COL  = OFF_ROW + (size_t)NROWS*4;  // f32 [N] col expsum
static constexpr size_t OFF_DIAG = OFF_COL + (size_t)NROWS*4;  // f32 [N] diagonal logits
static constexpr size_t OFF_ACC  = OFF_DIAG+ (size_t)NROWS*4;  // f32 [64] accumulators
static constexpr size_t OFF_AF   = OFF_ACC + 256;              // f32 [S,H] raw sampled rows
static constexpr size_t OFF_AFN  = OFF_AF  + (size_t)SSAMP*HDIM*4;
static constexpr size_t OFF_TE   = OFF_AFN + (size_t)SSAMP*HDIM*4; // f32 [POOL,H]
static constexpr size_t OFF_TEN  = OFF_TE  + (size_t)VPOOL*HDIM*4;
static constexpr size_t OFF_TL   = OFF_TEN + (size_t)VPOOL*HDIM*4; // f32 [S,POOL]

// ---------------------------------------------------------------------------
// Small helpers
// ---------------------------------------------------------------------------
__device__ inline float block_reduce_sum256(float v, float* sbuf) {
    int t = threadIdx.x;
    sbuf[t] = v; __syncthreads();
    #pragma unroll
    for (int s = 128; s > 0; s >>= 1) {
        if (t < s) sbuf[t] += sbuf[t + s];
        __syncthreads();
    }
    float r = sbuf[0]; __syncthreads();
    return r;
}

__device__ inline float gelu_exact(float x) {
    return 0.5f * x * (1.0f + erff(x * 0.70710678118654752f));
}

__device__ inline v8f wmma_bf16(v16bf a, v16bf b, v8f c) {
    return __builtin_amdgcn_wmma_f32_16x16x32_bf16(false, a, false, b, (short)0, c, false, false);
}

// A fragment (16x32 bf16): lanes 0-15 hold K=0..15, 16-31 K=16..31 (matches B)
__device__ inline v16bf load_frag_A(const __bf16* sA, int ldA, int m0, int lane) {
    int m    = m0 + (lane & 15);
    int half = lane >> 4;
    v16bf f;
    #pragma unroll
    for (int v = 0; v < 8; ++v) {
        int k = 2*v + 16*half;
        f[2*v]   = sA[m*ldA + k];
        f[2*v+1] = sA[m*ldA + k + 1];
    }
    return f;
}

// B fragment from a K-major [k][n] tile: lanes 0-15 hold K=0..15, 16-31 K=16..31
__device__ inline v16bf load_frag_B(const __bf16* sB, int ldB, int n0, int lane) {
    int n    = n0 + (lane & 15);
    int half = lane >> 4;
    v16bf f;
    #pragma unroll
    for (int v = 0; v < 8; ++v) {
        int k = 2*v + 16*half;
        f[2*v]   = sB[(k)   * ldB + n];
        f[2*v+1] = sB[(k+1) * ldB + n];
    }
    return f;
}

// B fragment from a row-major [n][k] tile (transpose done in the gather;
// per-lane reads are contiguous in k)
__device__ inline v16bf load_frag_B_rows(const __bf16* sBr, int ldB, int n0, int lane) {
    int n    = n0 + (lane & 15);
    int half = lane >> 4;
    v16bf f;
    #pragma unroll
    for (int v = 0; v < 8; ++v) {
        int k = 2*v + 16*half;
        f[2*v]   = sBr[n*ldB + k];
        f[2*v+1] = sBr[n*ldB + k + 1];
    }
    return f;
}

// ---------------------------------------------------------------------------
// Zero a float region
// ---------------------------------------------------------------------------
__global__ void k_zero(float* p, int n) {
    for (int i = blockIdx.x * blockDim.x + threadIdx.x; i < n; i += gridDim.x * blockDim.x)
        p[i] = 0.0f;
}

// ---------------------------------------------------------------------------
// f32 -> bf16 convert
// ---------------------------------------------------------------------------
__global__ void k_f2bf(const float* __restrict__ src, __bf16* __restrict__ dst, int n) {
    int i = blockIdx.x * blockDim.x + threadIdx.x;
    if (i < n) dst[i] = (__bf16)src[i];
}

// ---------------------------------------------------------------------------
// Row LayerNorm (+optional additive noise) -> bf16. One block per row.
// ---------------------------------------------------------------------------
__global__ __launch_bounds__(256) void k_ln_rows(const float* __restrict__ x,
                                                 const float* __restrict__ noise, int use_noise,
                                                 const float* __restrict__ g, const float* __restrict__ b,
                                                 __bf16* __restrict__ out) {
    __shared__ float sbuf[256];
    const int row = blockIdx.x;
    const int t   = threadIdx.x;
    float vloc[3];
    float s = 0.0f;
    #pragma unroll
    for (int i = 0; i < 3; ++i) {
        int h = t + i * 256;
        float v = x[(size_t)row * HDIM + h];
        if (use_noise) v += noise[(size_t)row * HDIM + h];
        vloc[i] = v;
        s += v;
    }
    float mean = block_reduce_sum256(s, sbuf) * (1.0f / HDIM);
    float sq = 0.0f;
    #pragma unroll
    for (int i = 0; i < 3; ++i) { float d = vloc[i] - mean; sq += d * d; }
    float var  = block_reduce_sum256(sq, sbuf) * (1.0f / HDIM);
    float rstd = rsqrtf(var + 1e-5f);
    #pragma unroll
    for (int i = 0; i < 3; ++i) {
        int h = t + i * 256;
        out[(size_t)row * HDIM + h] = (__bf16)((vloc[i] - mean) * rstd * g[h] + b[h]);
    }
}

// ---------------------------------------------------------------------------
// Row l2-normalize f32 -> bf16. One block per row.
// ---------------------------------------------------------------------------
__global__ __launch_bounds__(256) void k_rownorm_bf16(const float* __restrict__ u,
                                                      __bf16* __restrict__ out) {
    __shared__ float sbuf[256];
    const int row = blockIdx.x;
    const int t   = threadIdx.x;
    float sq = 0.0f;
    #pragma unroll
    for (int i = 0; i < 3; ++i) {
        float v = u[(size_t)row * HDIM + t + i * 256];
        sq += v * v;
    }
    float nrm = sqrtf(block_reduce_sum256(sq, sbuf));
    float inv = 1.0f / fmaxf(nrm, 1e-12f);
    #pragma unroll
    for (int i = 0; i < 3; ++i) {
        int h = t + i * 256;
        out[(size_t)row * HDIM + h] = (__bf16)(u[(size_t)row * HDIM + h] * inv);
    }
}

// ---------------------------------------------------------------------------
// Generic bf16 WMMA GEMM: C[M,N] = act(A[M,K] @ W[K,N] + bias[N])
// ACT==1: GELU, bf16 output; ACT==0: identity, f32 output.
// Block tile 128x128, BK=32, 8 waves (2x4), wave tile 64x32.
// A/B tiles staged with GLOBAL_LOAD_ASYNC_TO_LDS when available.
// ---------------------------------------------------------------------------
template<int ACT>
__global__ __launch_bounds__(256) void k_gemm_xw(const __bf16* __restrict__ A,
                                                 const __bf16* __restrict__ W,
                                                 const float* __restrict__ bias,
                                                 float* __restrict__ Cf,
                                                 __bf16* __restrict__ Cb,
                                                 int M, int K, int N) {
    __shared__ __bf16 sA[128 * 32];
    __shared__ __bf16 sB[32 * 128];
    const int tid  = threadIdx.x;
    const int lane = tid & 31;
    const int wave = tid >> 5;
    const int wM   = wave >> 2;       // 0..1
    const int wN   = wave & 3;        // 0..3
    const int bm   = blockIdx.x * 128;
    const int bn   = blockIdx.y * 128;

    v8f acc[4][2];
    #pragma unroll
    for (int mi = 0; mi < 4; ++mi)
        #pragma unroll
        for (int ni = 0; ni < 2; ++ni)
            #pragma unroll
            for (int r = 0; r < 8; ++r) acc[mi][ni][r] = 0.0f;

    for (int k0 = 0; k0 < K; k0 += 32) {
        // stage A tile (128x32): 512 16B-chunks, 2 per thread, direct to LDS
        #pragma unroll
        for (int i = 0; i < 2; ++i) {
            int vi = tid + i * 256;
            int r  = vi >> 2;
            int c  = (vi & 3) * 8;
            lds_copy16(&A[(size_t)(bm + r) * K + k0 + c], &sA[r * 32 + c]);
        }
        // stage B tile (32x128), K-major as stored in W
        #pragma unroll
        for (int i = 0; i < 2; ++i) {
            int vi = tid + i * 256;
            int r  = vi >> 4;
            int c  = (vi & 15) * 8;
            lds_copy16(&W[(size_t)(k0 + r) * N + bn + c], &sB[r * 128 + c]);
        }
        if (k0 + 32 < K) {   // prefetch next tiles into GL2
            __builtin_prefetch(&A[(size_t)(bm + (tid >> 1)) * K + k0 + 32], 0, 0);
            __builtin_prefetch(&W[(size_t)(k0 + 32 + (tid >> 3)) * N + bn], 0, 0);
        }
        lds_copy_wait();
        __syncthreads();

        v16bf afrag[4], bfrag[2];
        #pragma unroll
        for (int mi = 0; mi < 4; ++mi) afrag[mi] = load_frag_A(sA, 32, wM * 64 + mi * 16, lane);
        #pragma unroll
        for (int ni = 0; ni < 2; ++ni) bfrag[ni] = load_frag_B(sB, 128, wN * 32 + ni * 16, lane);
        #pragma unroll
        for (int mi = 0; mi < 4; ++mi)
            #pragma unroll
            for (int ni = 0; ni < 2; ++ni)
                acc[mi][ni] = wmma_bf16(afrag[mi], bfrag[ni], acc[mi][ni]);
        __syncthreads();
    }

    const int half = lane >> 4;
    #pragma unroll
    for (int mi = 0; mi < 4; ++mi) {
        #pragma unroll
        for (int ni = 0; ni < 2; ++ni) {
            int gn = bn + wN * 32 + ni * 16 + (lane & 15);
            float bv = bias[gn];
            #pragma unroll
            for (int r = 0; r < 8; ++r) {
                int gm = bm + wM * 64 + mi * 16 + r + 8 * half;
                float v = acc[mi][ni][r] + bv;
                if (ACT) {
                    Cb[(size_t)gm * N + gn] = (__bf16)gelu_exact(v);
                } else {
                    Cf[(size_t)gm * N + gn] = v;
                }
            }
        }
    }
}

// ---------------------------------------------------------------------------
// Contrastive logits: streaming exp-sums of (va @ vb^T) * 10 over rows & cols.
// Never materializes the 8192x8192 matrix. vb tile staged row-major [n][k]
// (contiguous async copies); transpose folded into the fragment gather.
// ---------------------------------------------------------------------------
__global__ __launch_bounds__(256) void k_gemm_logits(const __bf16* __restrict__ va,
                                                     const __bf16* __restrict__ vb,
                                                     float* __restrict__ rowsum,
                                                     float* __restrict__ colsum,
                                                     int N, int K) {
    __shared__ __bf16 sA[128 * 32];
    __shared__ __bf16 sBr[128 * 32];   // [n][k] row-major
    const int tid  = threadIdx.x;
    const int lane = tid & 31;
    const int wave = tid >> 5;
    const int wM   = wave >> 2;
    const int wN   = wave & 3;
    const int bm   = blockIdx.x * 128;
    const int bn   = blockIdx.y * 128;

    v8f acc[4][2];
    #pragma unroll
    for (int mi = 0; mi < 4; ++mi)
        #pragma unroll
        for (int ni = 0; ni < 2; ++ni)
            #pragma unroll
            for (int r = 0; r < 8; ++r) acc[mi][ni][r] = 0.0f;

    for (int k0 = 0; k0 < K; k0 += 32) {
        #pragma unroll
        for (int i = 0; i < 2; ++i) {
            int vi = tid + i * 256;
            int r  = vi >> 2;
            int c  = (vi & 3) * 8;
            lds_copy16(&va[(size_t)(bm + r) * K + k0 + c], &sA[r * 32 + c]);
            // same (r,c) mapping stages vb rows into [n][k]
            lds_copy16(&vb[(size_t)(bn + r) * K + k0 + c], &sBr[r * 32 + c]);
        }
        lds_copy_wait();
        __syncthreads();

        v16bf afrag[4], bfrag[2];
        #pragma unroll
        for (int mi = 0; mi < 4; ++mi) afrag[mi] = load_frag_A(sA, 32, wM * 64 + mi * 16, lane);
        #pragma unroll
        for (int ni = 0; ni < 2; ++ni) bfrag[ni] = load_frag_B_rows(sBr, 32, wN * 32 + ni * 16, lane);
        #pragma unroll
        for (int mi = 0; mi < 4; ++mi)
            #pragma unroll
            for (int ni = 0; ni < 2; ++ni)
                acc[mi][ni] = wmma_bf16(afrag[mi], bfrag[ni], acc[mi][ni]);
        __syncthreads();
    }

    // acc -> exp(10 * logit); bounded (|logit|<=10) so no max-stabilization needed
    #pragma unroll
    for (int mi = 0; mi < 4; ++mi)
        #pragma unroll
        for (int ni = 0; ni < 2; ++ni)
            #pragma unroll
            for (int r = 0; r < 8; ++r)
                acc[mi][ni][r] = __expf(acc[mi][ni][r] * TEMP_INV);

    const int half = lane >> 4;
    // row partial sums: reduce over the 16 lanes of each half
    #pragma unroll
    for (int mi = 0; mi < 4; ++mi) {
        #pragma unroll
        for (int r = 0; r < 8; ++r) {
            float v = acc[mi][0][r] + acc[mi][1][r];
            #pragma unroll
            for (int off = 1; off < 16; off <<= 1) v += __shfl_xor(v, off, 32);
            if ((lane & 15) == 0) {
                int gm = bm + wM * 64 + mi * 16 + r + 8 * half;
                atomicAdd(&rowsum[gm], v);
            }
        }
    }
    // column partial sums: reduce over r, mi, then across halves
    #pragma unroll
    for (int ni = 0; ni < 2; ++ni) {
        float v = 0.0f;
        #pragma unroll
        for (int mi = 0; mi < 4; ++mi)
            #pragma unroll
            for (int r = 0; r < 8; ++r) v += acc[mi][ni][r];
        v += __shfl_xor(v, 16, 32);
        if (half == 0) {
            int gn = bn + wN * 32 + ni * 16 + (lane & 15);
            atomicAdd(&colsum[gn], v);
        }
    }
}

// ---------------------------------------------------------------------------
// diag[i] = 10 * dot(va[i], vb[i])  (one block per row)
// ---------------------------------------------------------------------------
__global__ __launch_bounds__(256) void k_diag(const __bf16* __restrict__ va,
                                              const __bf16* __restrict__ vb,
                                              float* __restrict__ diag) {
    __shared__ float sbuf[256];
    const int row = blockIdx.x;
    float s = 0.0f;
    #pragma unroll
    for (int i = 0; i < 3; ++i) {
        int h = threadIdx.x + i * 256;
        s += (float)va[(size_t)row * HDIM + h] * (float)vb[(size_t)row * HDIM + h];
    }
    float tot = block_reduce_sum256(s, sbuf);
    if (threadIdx.x == 0) diag[row] = tot * TEMP_INV;
}

// ---------------------------------------------------------------------------
// con_loss = 0.5*(mean_i(log rowsum_i - diag_i) + mean_i(log colsum_i - diag_i))
// ---------------------------------------------------------------------------
__global__ __launch_bounds__(256) void k_con_finalize(const float* __restrict__ rowsum,
                                                      const float* __restrict__ colsum,
                                                      const float* __restrict__ diag,
                                                      float* __restrict__ out, int N) {
    __shared__ float sbuf[256];
    float sa = 0.0f, sb = 0.0f;
    for (int i = threadIdx.x; i < N; i += 256) {
        float d = diag[i];
        sa += __logf(rowsum[i]) - d;
        sb += __logf(colsum[i]) - d;
    }
    float ra = block_reduce_sum256(sa, sbuf);
    float rb = block_reduce_sum256(sb, sbuf);
    if (threadIdx.x == 0) out[0] = 0.5f * (ra / N + rb / N);
}

// ---------------------------------------------------------------------------
// Compensation elementwise + comp_reg partial sums
// ---------------------------------------------------------------------------
__global__ __launch_bounds__(256) void k_compensate(const float* __restrict__ latent,
                                                    const float* __restrict__ shead,
                                                    const float* __restrict__ bhead,
                                                    float* __restrict__ aligned,
                                                    float* __restrict__ scaleO,
                                                    float* __restrict__ biasO,
                                                    float* __restrict__ accS,
                                                    float* __restrict__ accB) {
    __shared__ float sbuf[256];
    const size_t base   = (size_t)blockIdx.x * 256 + threadIdx.x;
    const size_t stride = (size_t)gridDim.x * 256;
    float ls = 0.0f, lb = 0.0f;
    #pragma unroll
    for (int i = 0; i < 4; ++i) {
        size_t j = base + (size_t)i * stride;
        float sc = 1.0f + LIM * tanhf(shead[j]);
        float bi = LIM * tanhf(bhead[j]);
        aligned[j] = sc * latent[j] + bi;
        scaleO[j]  = sc;
        biasO[j]   = bi;
        float d = sc - 1.0f;
        ls += d * d;
        lb += bi * bi;
    }
    float rs = block_reduce_sum256(ls, sbuf);
    float rb = block_reduce_sum256(lb, sbuf);
    if (threadIdx.x == 0) { atomicAdd(accS, rs); atomicAdd(accB, rb); }
}

// ---------------------------------------------------------------------------
// Gather rows by index + l2-normalize; writes raw and normalized f32
// ---------------------------------------------------------------------------
__global__ __launch_bounds__(256) void k_gather_norm(const float* __restrict__ src,
                                                     const int* __restrict__ idx,
                                                     float* __restrict__ raw,
                                                     float* __restrict__ nrm) {
    __shared__ float sbuf[256];
    const int r   = blockIdx.x;
    const int row = idx[r];
    float sq = 0.0f;
    #pragma unroll
    for (int i = 0; i < 3; ++i) {
        int h = threadIdx.x + i * 256;
        float v = src[(size_t)row * HDIM + h];
        raw[(size_t)r * HDIM + h] = v;
        sq += v * v;
    }
    float n   = sqrtf(block_reduce_sum256(sq, sbuf));
    float inv = 1.0f / fmaxf(n, 1e-12f);
    #pragma unroll
    for (int i = 0; i < 3; ++i) {
        int h = threadIdx.x + i * 256;
        nrm[(size_t)r * HDIM + h] = raw[(size_t)r * HDIM + h] * inv;
    }
}

// ---------------------------------------------------------------------------
// tl[s,p] = 10 * dot(afn[s], ten[p]);  one block per sample s
// ---------------------------------------------------------------------------
__global__ __launch_bounds__(256) void k_tl(const float* __restrict__ afn,
                                            const float* __restrict__ ten,
                                            float* __restrict__ tl) {
    __shared__ float arow[HDIM];
    const int s = blockIdx.x;
    for (int h = threadIdx.x; h < HDIM; h += 256) arow[h] = afn[(size_t)s * HDIM + h];
    __syncthreads();
    #pragma unroll
    for (int pi = 0; pi < 4; ++pi) {
        int p = threadIdx.x + pi * 256;
        float d = 0.0f;
        for (int h = 0; h < HDIM; ++h) d += arow[h] * ten[(size_t)p * HDIM + h];
        tl[(size_t)s * VPOOL + p] = d * TEMP_INV;
    }
}

// ---------------------------------------------------------------------------
// Top-64 of 1024, softmax weights, prototype, MSE partial. One block per s.
// ---------------------------------------------------------------------------
__global__ __launch_bounds__(256) void k_topk_proto(const float* __restrict__ tl,
                                                    const float* __restrict__ te,
                                                    const float* __restrict__ af,
                                                    float* __restrict__ accP) {
    __shared__ float vals[VPOOL];
    __shared__ float rv[256];
    __shared__ int   ri[256];
    __shared__ float topv[KTOP];
    __shared__ int   topi[KTOP];
    __shared__ float topw[KTOP];
    __shared__ float tsum;
    __shared__ float sbuf[256];
    const int s = blockIdx.x;
    const int t = threadIdx.x;

    for (int p = t; p < VPOOL; p += 256) vals[p] = tl[(size_t)s * VPOOL + p];
    __syncthreads();

    for (int it = 0; it < KTOP; ++it) {
        float lm = -1e30f; int li = 0;
        for (int p = t; p < VPOOL; p += 256) {
            float v = vals[p];
            if (v > lm) { lm = v; li = p; }
        }
        rv[t] = lm; ri[t] = li; __syncthreads();
        #pragma unroll
        for (int s2 = 128; s2 > 0; s2 >>= 1) {
            if (t < s2 && rv[t + s2] > rv[t]) { rv[t] = rv[t + s2]; ri[t] = ri[t + s2]; }
            __syncthreads();
        }
        if (t == 0) { topv[it] = rv[0]; topi[it] = ri[0]; vals[ri[0]] = -1e30f; }
        __syncthreads();
    }
    if (t < KTOP) topw[t] = __expf(topv[t] - topv[0]);
    __syncthreads();
    if (t == 0) {
        float ss = 0.0f;
        for (int k = 0; k < KTOP; ++k) ss += topw[k];
        tsum = ss;
    }
    __syncthreads();
    float winv = 1.0f / tsum;

    float loc = 0.0f;
    #pragma unroll
    for (int i = 0; i < 3; ++i) {
        int h = t + i * 256;
        float proto = 0.0f;
        for (int k = 0; k < KTOP; ++k)
            proto += topw[k] * te[(size_t)topi[k] * HDIM + h];
        proto *= winv;
        float d = af[(size_t)s * HDIM + h] - proto;
        loc += d * d;
    }
    float tot = block_reduce_sum256(loc, sbuf);
    if (t == 0) atomicAdd(accP, tot);
}

// ---------------------------------------------------------------------------
// Column moments (mean/std, ddof=0) of af[256,H] vs te[1024,H]. Block per h.
// ---------------------------------------------------------------------------
__global__ __launch_bounds__(256) void k_moments(const float* __restrict__ af,
                                                 const float* __restrict__ te,
                                                 float* __restrict__ accM1,
                                                 float* __restrict__ accM2) {
    __shared__ float sbuf[256];
    const int h = blockIdx.x;
    const int t = threadIdx.x;
    float av = af[(size_t)t * HDIM + h];
    float as = av, aq = av * av;
    float ts = 0.0f, tq = 0.0f;
    #pragma unroll
    for (int i = 0; i < 4; ++i) {
        float v = te[(size_t)(t + i * 256) * HDIM + h];
        ts += v; tq += v * v;
    }
    float asum = block_reduce_sum256(as, sbuf);
    float asq  = block_reduce_sum256(aq, sbuf);
    float tsum = block_reduce_sum256(ts, sbuf);
    float tsq  = block_reduce_sum256(tq, sbuf);
    if (t == 0) {
        float am = asum / SSAMP;
        float avar = fmaxf(asq / SSAMP - am * am, 0.0f);
        float tm = tsum / VPOOL;
        float tvar = fmaxf(tsq / VPOOL - tm * tm, 0.0f);
        float dm = am - tm;
        float ds = sqrtf(avar) - sqrtf(tvar);
        atomicAdd(accM1, dm * dm);
        atomicAdd(accM2, ds * ds);
    }
}

// ---------------------------------------------------------------------------
// Final scalar losses
// acc[0]=sum(scale-1)^2 acc[1]=sum bias^2 acc[2]=sum proto sq err
// acc[3]=sum mean diffs  acc[4]=sum std diffs
// ---------------------------------------------------------------------------
__global__ void k_finalize(const float* __restrict__ acc,
                           float* __restrict__ tokenOut,
                           float* __restrict__ compOut) {
    tokenOut[0] = acc[2] / (float)(SSAMP * HDIM)
                + MW * (acc[3] / (float)HDIM + acc[4] / (float)HDIM);
    compOut[0]  = acc[0] / (float)NHs + acc[1] / (float)NHs;
}

// ---------------------------------------------------------------------------
// Host launch
// ---------------------------------------------------------------------------
extern "C" void kernel_launch(void* const* d_in, const int* in_sizes, int n_in,
                              void* d_out, int out_size, void* d_ws, size_t ws_size,
                              hipStream_t stream) {
    (void)in_sizes; (void)n_in; (void)out_size; (void)ws_size;
    const float* latent  = (const float*)d_in[0];
    const float* tokemb  = (const float*)d_in[1];
    const float* ln_g[3] = {(const float*)d_in[2],  (const float*)d_in[8],  (const float*)d_in[14]};
    const float* ln_b[3] = {(const float*)d_in[3],  (const float*)d_in[9],  (const float*)d_in[15]};
    const float* w1[3]   = {(const float*)d_in[4],  (const float*)d_in[10], (const float*)d_in[16]};
    const float* b1[3]   = {(const float*)d_in[5],  (const float*)d_in[11], (const float*)d_in[17]};
    const float* w2[3]   = {(const float*)d_in[6],  (const float*)d_in[12], (const float*)d_in[18]};
    const float* b2[3]   = {(const float*)d_in[7],  (const float*)d_in[13], (const float*)d_in[19]};
    const float* noise_a = (const float*)d_in[20];
    const float* noise_b = (const float*)d_in[21];
    const int*   samp    = (const int*)d_in[22];
    const int*   toki    = (const int*)d_in[23];

    char* ws = (char*)d_ws;
    __bf16* XN  = (__bf16*)(ws + OFF_XN);
    __bf16* H1  = (__bf16*)(ws + OFF_H1);
    __bf16* WB  = (__bf16*)(ws + OFF_WB);
    float*  HA  = (float*)(ws + OFF_HA);
    float*  HB  = (float*)(ws + OFF_HB);
    __bf16* VA  = (__bf16*)(ws + OFF_VA);
    __bf16* VB  = (__bf16*)(ws + OFF_VB);
    float*  ROW = (float*)(ws + OFF_ROW);
    float*  COL = (float*)(ws + OFF_COL);
    float*  DIA = (float*)(ws + OFF_DIAG);
    float*  ACC = (float*)(ws + OFF_ACC);
    float*  AF  = (float*)(ws + OFF_AF);
    float*  AFN = (float*)(ws + OFF_AFN);
    float*  TE  = (float*)(ws + OFF_TE);
    float*  TEN = (float*)(ws + OFF_TEN);
    float*  TL  = (float*)(ws + OFF_TL);

    float* out      = (float*)d_out;
    float* alignedO = out;
    float* conO     = out + NHs;
    float* tokenO   = out + NHs + 1;
    float* compO    = out + NHs + 2;
    float* scaleO   = out + NHs + 3;
    float* biasO    = out + NHs + 3 + NHs;

    const int WN = 589824; // 768*768
    const dim3 gemmGrid(NROWS / 128, HDIM / 128);
    const dim3 logitsGrid(NROWS / 128, NROWS / 128);

    // 0) zero rowsum/colsum/diag/accumulators
    {
        int nz = (int)((OFF_ACC + 256 - OFF_ROW) / 4);
        k_zero<<<64, 256, 0, stream>>>(ROW, nz);
    }

    // helper: one head MLP pass: LN(x)+opt noise -> GEMM1(GELU) -> GEMM2 -> dst f32
    auto head_pass = [&](const float* x, const float* noise, int use_noise, int hi, float* dst) {
        k_ln_rows<<<NROWS, 256, 0, stream>>>(x, noise, use_noise, ln_g[hi], ln_b[hi], XN);
        k_f2bf<<<WN / 256, 256, 0, stream>>>(w1[hi], WB, WN);
        k_gemm_xw<1><<<gemmGrid, 256, 0, stream>>>(XN, WB, b1[hi], nullptr, H1,
                                                   NROWS, HDIM, HDIM);
        k_f2bf<<<WN / 256, 256, 0, stream>>>(w2[hi], WB, WN);
        k_gemm_xw<0><<<gemmGrid, 256, 0, stream>>>(H1, WB, b2[hi], dst, nullptr,
                                                   NROWS, HDIM, HDIM);
    };

    // 1) scale & bias heads on latent
    head_pass(latent, nullptr, 0, 0, HA);
    head_pass(latent, nullptr, 0, 1, HB);

    // 2) compensation: aligned/scale/bias + comp_reg partials
    k_compensate<<<NROWS * HDIM / 1024, 256, 0, stream>>>(latent, HA, HB,
                                                          alignedO, scaleO, biasO,
                                                          &ACC[0], &ACC[1]);

    // 3) contrastive views through con head, then l2-normalize to bf16
    head_pass(alignedO, noise_a, 1, 2, HA);
    k_rownorm_bf16<<<NROWS, 256, 0, stream>>>(HA, VA);
    head_pass(alignedO, noise_b, 1, 2, HB);
    k_rownorm_bf16<<<NROWS, 256, 0, stream>>>(HB, VB);

    // 4) contrastive loss: diagonal + streamed exp-sums + finalize
    k_diag<<<NROWS, 256, 0, stream>>>(VA, VB, DIA);
    k_gemm_logits<<<logitsGrid, 256, 0, stream>>>(VA, VB, ROW, COL, NROWS, HDIM);
    k_con_finalize<<<1, 256, 0, stream>>>(ROW, COL, DIA, conO, NROWS);

    // 5) token distribution loss
    k_gather_norm<<<SSAMP, 256, 0, stream>>>(alignedO, samp, AF, AFN);
    k_gather_norm<<<VPOOL, 256, 0, stream>>>(tokemb, toki, TE, TEN);
    k_tl<<<SSAMP, 256, 0, stream>>>(AFN, TEN, TL);
    k_topk_proto<<<SSAMP, 256, 0, stream>>>(TL, TE, AF, &ACC[2]);
    k_moments<<<HDIM, 256, 0, stream>>>(AF, TE, &ACC[3], &ACC[4]);

    // 6) scalar finalizers
    k_finalize<<<1, 1, 0, stream>>>(ACC, tokenO, compO);
}
